// GraphNetwork_56624848831081
// MI455X (gfx1250) — matrix operand order repaired and checked by reading, not compiled, not run on hardware
//
#include <hip/hip_runtime.h>
#include <math.h>

#define LEAKY_SLOPE 0.01f
#define BN_EPS 1e-5f
#define NN 100
#define N2 10000

typedef __attribute__((ext_vector_type(16))) __bf16 v16bf;
typedef __attribute__((ext_vector_type(8)))  float v8f;

union Frag { unsigned u[8]; v16bf v; };

__device__ __forceinline__ unsigned short f2bf(float f) {
  unsigned u = __float_as_uint(f);
  unsigned r = 0x7FFFu + ((u >> 16) & 1u);   // round to nearest even
  return (unsigned short)((u + r) >> 16);
}
__device__ __forceinline__ float bf2f(unsigned short h) {
  return __uint_as_float(((unsigned)h) << 16);
}
__device__ __forceinline__ float leakyf(float t) { return t >= 0.f ? t : LEAKY_SLOPE * t; }

// ---- async global->LDS copy (CDNA5), with synchronous fallback ------------
typedef int v2i __attribute__((vector_size(8)));
typedef v2i __attribute__((address_space(1))) as1v2i;
typedef v2i __attribute__((address_space(3))) as3v2i;

__device__ __forceinline__ void copy_b64_g2l(const void* g, void* l) {
#if defined(__AMDGCN__) && __has_builtin(__builtin_amdgcn_global_load_async_to_lds_b64)
  void* gnc = const_cast<void*>(g);
  __builtin_amdgcn_global_load_async_to_lds_b64((as1v2i*)gnc, (as3v2i*)l, 0, 0);
#else
  *(unsigned long long*)l = *(const unsigned long long*)g;
#endif
}
__device__ __forceinline__ void wait_async_done() {
#if defined(__AMDGCN__) && __has_builtin(__builtin_amdgcn_global_load_async_to_lds_b64)
#if __has_builtin(__builtin_amdgcn_s_wait_asynccnt)
  __builtin_amdgcn_s_wait_asynccnt(0);
#else
  asm volatile("s_wait_asynccnt 0" ::: "memory");
#endif
#endif
}

// ---------------------------------------------------------------------------
// Split fp32 weight [d x twoD] into transposed bf16 hi/lo [twoD x d]
// ---------------------------------------------------------------------------
__global__ void split_w_kernel(const float* __restrict__ w,
                               unsigned short* __restrict__ whi,
                               unsigned short* __restrict__ wlo,
                               int d, int twoD) {
  long long idx = (long long)blockIdx.x * blockDim.x + threadIdx.x;
  long long tot = (long long)d * twoD;
  if (idx >= tot) return;
  int k = (int)(idx % d);
  int c = (int)(idx / d);
  float v = w[(size_t)k * twoD + c];
  unsigned short hi = f2bf(v);
  unsigned short lo = f2bf(v - bf2f(hi));
  whi[idx] = hi;
  wlo[idx] = lo;
}

// ---------------------------------------------------------------------------
// Precompute A = |x_i - x_j| split to bf16 hi/lo, packed u32 pairs.
// Layout: ahi[(i*100+j)*(d/2) + k2] holds bf16(k=2*k2), bf16(k=2*k2+1).
// ---------------------------------------------------------------------------
__global__ void precompute_A_kernel(const float* __restrict__ x,
                                    unsigned* __restrict__ ahi,
                                    unsigned* __restrict__ alo, int d) {
  int idx = blockIdx.x * 256 + threadIdx.x;
  int dh = d >> 1;
  if (idx >= N2 * dh) return;
  int r = idx / dh, k2 = idx % dh;
  int i = r / NN, j = r % NN;
  const float2 xi = *(const float2*)(x + (size_t)i * d + 2 * k2);
  const float2 xj = *(const float2*)(x + (size_t)j * d + 2 * k2);
  float a0 = fabsf(xi.x - xj.x);
  float a1 = fabsf(xi.y - xj.y);
  unsigned short h0 = f2bf(a0), h1 = f2bf(a1);
  unsigned short l0 = f2bf(a0 - bf2f(h0)), l1 = f2bf(a1 - bf2f(h1));
  ahi[idx] = (unsigned)h0 | ((unsigned)h1 << 16);
  alo[idx] = (unsigned)l0 | ((unsigned)l1 << 16);
}

// ---------------------------------------------------------------------------
// Big GEMM: h[(i*100+j), c] = sum_k A[(i,j),k] * w1[k,c] + b1[c]
// Grid: (twoD/128, 100), block 256 (8 waves). bf16 split (3 WMMAs / K32).
// Staging is pure global->LDS (async on CDNA5).
// ---------------------------------------------------------------------------
__global__ __launch_bounds__(256) void gemm_h_kernel(
    const unsigned* __restrict__ ahi_g,          // [10000 x d/2] u32 (bf16 pairs)
    const unsigned* __restrict__ alo_g,
    const unsigned short* __restrict__ w1t_hi,   // [twoD x d] bf16
    const unsigned short* __restrict__ w1t_lo,
    const float* __restrict__ b1,
    float* __restrict__ h,
    int d, int twoD) {
  // LDS: A (112 rows x 64 k, stride 68 bf16 = 136 B), W (128 cols x 64 k)
  __shared__ unsigned AhiU[112 * 34];
  __shared__ unsigned AloU[112 * 34];
  __shared__ unsigned WhiU[128 * 34];
  __shared__ unsigned WloU[128 * 34];

  const int t    = threadIdx.x;
  const int lane = t & 31;
  const int wv   = t >> 5;          // wave id 0..7 -> its 16-col tile
  const int i    = blockIdx.y;      // node i
  const int c0   = blockIdx.x * 128;
  const int half = lane >> 4;
  const int nl   = lane & 15;

  v8f acc[7];
  const v8f vzero = {0.f, 0.f, 0.f, 0.f, 0.f, 0.f, 0.f, 0.f};
#pragma unroll
  for (int m = 0; m < 7; ++m) acc[m] = vzero;

  const char* ahiB = (const char*)ahi_g;
  const char* aloB = (const char*)alo_g;
  const char* whiB = (const char*)w1t_hi;
  const char* wloB = (const char*)w1t_lo;

  for (int kc = 0; kc < d; kc += 64) {
    __syncthreads();   // previous chunk fully consumed before LDS overwrite
    // ---- stage A tile: 112 rows x 64 B, b64 granularity (896 qwords) ----
#pragma unroll
    for (int it = 0; it < 4; ++it) {
      int idx = t + it * 256;
      if (idx < 896) {
        int j = idx >> 3, q = idx & 7;              // q = 8B chunk in row
        int jc = j < NN ? j : NN - 1;               // clamp pad rows
        size_t goff = ((size_t)(i * NN + jc) * d + kc) * 2 + (size_t)q * 8;
        int loff = j * 136 + q * 8;
        copy_b64_g2l(ahiB + goff, (char*)AhiU + loff);
        copy_b64_g2l(aloB + goff, (char*)AloU + loff);
      }
    }
    // ---- stage W tile: 128 cols x 64 B (1024 qwords) ----
#pragma unroll
    for (int it = 0; it < 4; ++it) {
      int idx = t + it * 256;
      int c = idx >> 3, q = idx & 7;
      size_t goff = ((size_t)(c0 + c) * d + kc) * 2 + (size_t)q * 8;
      int loff = c * 136 + q * 8;
      copy_b64_g2l(whiB + goff, (char*)WhiU + loff);
      copy_b64_g2l(wloB + goff, (char*)WloU + loff);
    }
    wait_async_done();
    __syncthreads();

#pragma unroll
    for (int kb = 0; kb < 2; ++kb) {              // two K=32 blocks
      Frag bhi, blo;
      {
        int cb = wv * 16 + nl;                    // column for this lane
        int base = cb * 34 + kb * 16 + half * 4;  // B: K = 16g + 8*half + 2rr
#pragma unroll
        for (int r = 0; r < 8; ++r) {
          bhi.u[r] = WhiU[base + (r >> 2) * 8 + (r & 3)];
          blo.u[r] = WloU[base + (r >> 2) * 8 + (r & 3)];
        }
      }
#pragma unroll
      for (int mt = 0; mt < 7; ++mt) {
        Frag ahi, alo;
        int j = mt * 16 + nl;                     // row for this lane
        int base = j * 34 + kb * 16 + half * 4;   // A: K = 16g + 8*half + 2rr
#pragma unroll
        for (int r = 0; r < 8; ++r) {
          int g = r >> 2, rr = r & 3;
          ahi.u[r] = AhiU[base + g * 8 + rr];
          alo.u[r] = AloU[base + g * 8 + rr];
        }
        acc[mt] = __builtin_amdgcn_wmma_f32_16x16x32_bf16(
            false, ahi.v, false, bhi.v, (short)0, acc[mt], false, false);
        acc[mt] = __builtin_amdgcn_wmma_f32_16x16x32_bf16(
            false, ahi.v, false, blo.v, (short)0, acc[mt], false, false);
        acc[mt] = __builtin_amdgcn_wmma_f32_16x16x32_bf16(
            false, alo.v, false, bhi.v, (short)0, acc[mt], false, false);
      }
    }
  }

  // ---- epilogue: fold bias, store valid rows ----
  int c = c0 + wv * 16 + nl;
  float bias = b1[c];
#pragma unroll
  for (int mt = 0; mt < 7; ++mt) {
#pragma unroll
    for (int v = 0; v < 8; ++v) {
      int j = mt * 16 + v + 8 * half;             // C layout: m = v + 8*half
      if (j < NN) h[(size_t)(i * NN + j) * twoD + c] = acc[mt][v] + bias;
    }
  }
}

// ---------------------------------------------------------------------------
// Column partial sums/sumsq of h over 10000 rows (deterministic two-level).
// Grid: (twoD/256, 25); each y-block covers 400 rows.
// ---------------------------------------------------------------------------
__global__ void colpart_kernel(const float* __restrict__ h,
                               float* __restrict__ psum, float* __restrict__ psq,
                               int twoD) {
  int c = blockIdx.x * 256 + threadIdx.x;
  int by = blockIdx.y;
  float s = 0.f, q = 0.f;
  int r0 = by * 400;
  for (int r = r0; r < r0 + 400; ++r) {
    float v = h[(size_t)r * twoD + c];
    s += v;
    q += v * v;
  }
  psum[(size_t)by * twoD + c] = s;
  psq[(size_t)by * twoD + c]  = q;
}

__global__ void finstats_kernel(const float* __restrict__ psum,
                                const float* __restrict__ psq,
                                const float* __restrict__ g, const float* __restrict__ be,
                                float* __restrict__ alpha, float* __restrict__ beta,
                                int twoD) {
  int c = blockIdx.x * 256 + threadIdx.x;
  float s = 0.f, q = 0.f;
  for (int b = 0; b < 25; ++b) {
    s += psum[(size_t)b * twoD + c];
    q += psq[(size_t)b * twoD + c];
  }
  float m = s * (1.f / N2);
  float var = q * (1.f / N2) - m * m;
  float a = g[c] * rsqrtf(var + BN_EPS);
  alpha[c] = a;
  beta[c]  = be[c] - m * a;
}

// ---------------------------------------------------------------------------
// Per-row fused BN + leaky + (.)@w2 + sigmoid.  Grid: 10000 blocks x 256.
// ---------------------------------------------------------------------------
__global__ __launch_bounds__(256) void scores_kernel(
    const float* __restrict__ h, const float* __restrict__ alpha,
    const float* __restrict__ beta, const float* __restrict__ w2,
    const float* __restrict__ b2, float* __restrict__ s, int twoD) {
  __shared__ float red[256];
  int r = blockIdx.x, t = threadIdx.x;
  float p = 0.f;
  for (int c = t; c < twoD; c += 256) {
    float v = alpha[c] * h[(size_t)r * twoD + c] + beta[c];
    p += leakyf(v) * w2[c];
  }
  red[t] = p;
  __syncthreads();
  for (int off = 128; off > 0; off >>= 1) {
    if (t < off) red[t] += red[t + off];
    __syncthreads();
  }
  if (t == 0) {
    float z = red[0] + b2[0];
    s[r] = 1.f / (1.f + expf(-z));
  }
}

// dis row sums (off-diagonal), yscale[j] = 1 / sum_m dis[j,m]
__global__ void rowsum_kernel(const float* __restrict__ s, float* __restrict__ ysc) {
  int j = threadIdx.x;
  if (j >= NN) return;
  float sum = 0.f;
  for (int m = 0; m < NN; ++m)
    if (m != j) sum += s[j * NN + m];
  ysc[j] = 1.f / sum;
}

// agg[i,k] = sum_{j!=i} s[i,j] * ysc[j] * x[j,k]
__global__ void agg_kernel(const float* __restrict__ s, const float* __restrict__ ysc,
                           const float* __restrict__ x, float* __restrict__ agg, int d) {
  int k = blockIdx.x * 256 + threadIdx.x;
  int i = blockIdx.y;
  if (k >= d) return;
  float a = 0.f;
  for (int j = 0; j < NN; ++j) {
    if (j == i) continue;
    a += s[i * NN + j] * ysc[j] * x[(size_t)j * d + k];
  }
  agg[(size_t)i * d + k] = a;
}

// z[i,c] = nb[c] + cat(x,agg)[i,:] @ nw[:,c]
__global__ void node_kernel(const float* __restrict__ x, const float* __restrict__ agg,
                            const float* __restrict__ nw, const float* __restrict__ nb,
                            float* __restrict__ z, int d, int dn) {
  int c = blockIdx.x * 256 + threadIdx.x;
  int i = blockIdx.y;
  float acc = nb[c];
  for (int k = 0; k < d; ++k) acc += x[(size_t)i * d + k] * nw[(size_t)k * dn + c];
  for (int k = 0; k < d; ++k) acc += agg[(size_t)i * d + k] * nw[(size_t)(d + k) * dn + c];
  z[(size_t)i * dn + c] = acc;
}

// BN over 100-row batch + leaky -> new x
__global__ void bn_node_kernel(const float* __restrict__ z, const float* __restrict__ g,
                               const float* __restrict__ be, float* __restrict__ xo, int dn) {
  int c = blockIdx.x * 256 + threadIdx.x;
  float s = 0.f, q = 0.f;
  for (int r = 0; r < NN; ++r) {
    float v = z[(size_t)r * dn + c];
    s += v;
    q += v * v;
  }
  float m = s * (1.f / NN);
  float var = q * (1.f / NN) - m * m;
  float a = g[c] * rsqrtf(var + BN_EPS);
  float b = be[c] - m * a;
  for (int r = 0; r < NN; ++r)
    xo[(size_t)r * dn + c] = leakyf(a * z[(size_t)r * dn + c] + b);
}

__global__ void copy_kernel(const float* __restrict__ src, float* __restrict__ dst, int n) {
  int i = blockIdx.x * 256 + threadIdx.x;
  if (i < n) dst[i] = src[i];
}

// ---------------------------------------------------------------------------
extern "C" void kernel_launch(void* const* d_in, const int* in_sizes, int n_in,
                              void* d_out, int out_size, void* d_ws, size_t ws_size,
                              hipStream_t stream) {
  (void)in_sizes; (void)n_in; (void)out_size; (void)ws_size;

  const float* node_feat = (const float*)d_in[0];
  struct EP { const float *w1, *b1, *g, *be, *w2, *b2, *nw, *nb, *ng, *nbe; };
  EP L[3];
  for (int l = 0; l < 2; ++l) {
    int b = 1 + l * 10;   // ew1, eb1, eg, ebe, ew2, eb2, nw, nb, ng, nbe
    L[l].w1 = (const float*)d_in[b + 0]; L[l].b1 = (const float*)d_in[b + 1];
    L[l].g  = (const float*)d_in[b + 2]; L[l].be = (const float*)d_in[b + 3];
    L[l].w2 = (const float*)d_in[b + 4]; L[l].b2 = (const float*)d_in[b + 5];
    L[l].nw = (const float*)d_in[b + 6]; L[l].nb = (const float*)d_in[b + 7];
    L[l].ng = (const float*)d_in[b + 8]; L[l].nbe = (const float*)d_in[b + 9];
  }
  L[2].w1 = (const float*)d_in[21]; L[2].b1 = (const float*)d_in[22];
  L[2].g  = (const float*)d_in[23]; L[2].be = (const float*)d_in[24];
  L[2].w2 = (const float*)d_in[25]; L[2].b2 = (const float*)d_in[26];
  L[2].nw = nullptr; L[2].nb = nullptr; L[2].ng = nullptr; L[2].nbe = nullptr;

  // ---- workspace carve ----
  char* p = (char*)d_ws;
  auto alloc = [&](size_t bytes) {
    char* r = p;
    p += (bytes + 255) & ~(size_t)255;
    return r;
  };
  float*          hbuf = (float*)alloc((size_t)N2 * 2048 * 4);            // 82 MB
  unsigned*       aghi = (unsigned*)alloc((size_t)N2 * 1024 * 2);         // 20.5 MB
  unsigned*       aglo = (unsigned*)alloc((size_t)N2 * 1024 * 2);         // 20.5 MB
  unsigned short* whi  = (unsigned short*)alloc((size_t)2048 * 1024 * 2); // 4 MB
  unsigned short* wlo  = (unsigned short*)alloc((size_t)2048 * 1024 * 2); // 4 MB
  float*          psum = (float*)alloc((size_t)25 * 2048 * 4);
  float*          psq  = (float*)alloc((size_t)25 * 2048 * 4);
  float*          alp  = (float*)alloc(2048 * 4);
  float*          bet  = (float*)alloc(2048 * 4);
  float*          sbuf = (float*)alloc((size_t)N2 * 4);
  float*          ysc  = (float*)alloc(NN * 4);
  float*          aggb = (float*)alloc((size_t)NN * 1024 * 4);
  float*          zbuf = (float*)alloc((size_t)NN * 1024 * 4);
  float*          xA   = (float*)alloc((size_t)NN * 1024 * 4);
  float*          xB   = (float*)alloc((size_t)NN * 1024 * 4);
  float*          xbufs[2] = {xA, xB};

  const float* xcur = node_feat;
  int d = 640;

  for (int l = 0; l < 3; ++l) {
    int twoD = 2 * d;
    {
      long long tot = (long long)d * twoD;
      int blocks = (int)((tot + 255) / 256);
      split_w_kernel<<<blocks, 256, 0, stream>>>(L[l].w1, whi, wlo, d, twoD);
    }
    {
      int tot = N2 * (d >> 1);
      precompute_A_kernel<<<(tot + 255) / 256, 256, 0, stream>>>(xcur, aghi, aglo, d);
    }
    gemm_h_kernel<<<dim3(twoD / 128, NN), 256, 0, stream>>>(
        aghi, aglo, whi, wlo, L[l].b1, hbuf, d, twoD);
    colpart_kernel<<<dim3(twoD / 256, 25), 256, 0, stream>>>(hbuf, psum, psq, twoD);
    finstats_kernel<<<twoD / 256, 256, 0, stream>>>(psum, psq, L[l].g, L[l].be,
                                                    alp, bet, twoD);
    float* sout = (l == 2) ? ((float*)d_out + NN * 1024) : sbuf;
    scores_kernel<<<N2, 256, 0, stream>>>(hbuf, alp, bet, L[l].w2, L[l].b2, sout, twoD);
    if (l == 2) break;

    rowsum_kernel<<<1, 128, 0, stream>>>(sbuf, ysc);
    agg_kernel<<<dim3((d + 255) / 256, NN), 256, 0, stream>>>(sbuf, ysc, xcur, aggb, d);
    int dn = 1024;
    node_kernel<<<dim3(dn / 256, NN), 256, 0, stream>>>(xcur, aggb, L[l].nw, L[l].nb,
                                                        zbuf, d, dn);
    bn_node_kernel<<<dn / 256, 256, 0, stream>>>(zbuf, L[l].ng, L[l].nbe, xbufs[l & 1], dn);
    xcur = xbufs[l & 1];
    d = dn;
  }

  copy_kernel<<<(NN * 1024 + 255) / 256, 256, 0, stream>>>(xcur, (float*)d_out, NN * 1024);
}
